// Atten_Cross_8100308321063
// MI455X (gfx1250) — compile-verified, matrix operand order checked by reading, not run on hardware
//
#include <hip/hip_runtime.h>
#include <hip/hip_bf16.h>

// Atten_Cross: y = gamma * (softmax(Q Kt) V) + x
// Flash-attention on v_wmma_f32_16x16x32_bf16 with TDM (tensor_load_to_lds)
// double-buffered K/V tile staging. Workspace requirement: >= 40 MiB.

#define BB   8
#define CC   512
#define NN   4096
#define CQKD 64

typedef __attribute__((ext_vector_type(16))) __bf16 bf16x16;
typedef __attribute__((ext_vector_type(8)))  float  f32x8;
typedef __attribute__((ext_vector_type(4)))  unsigned u32x4;
typedef __attribute__((ext_vector_type(8)))  unsigned u32x8;

// ---------------- Phase 1: 1x1-conv projections -> bf16 Q/K/V ----------------
#define P1_PIX 16
__global__ __launch_bounds__(256) void proj_kernel(
    const float* __restrict__ x, const float* __restrict__ cond,
    const float* __restrict__ wq, const float* __restrict__ bq,
    const float* __restrict__ wk, const float* __restrict__ bk,
    const float* __restrict__ wv, const float* __restrict__ bv,
    __bf16* __restrict__ Qb, __bf16* __restrict__ Kb, __bf16* __restrict__ Vb)
{
  __shared__ float xs[CC][P1_PIX];
  __shared__ float cs[CC][P1_PIX];
  const int blk = blockIdx.x;
  const int b  = blk / (NN / P1_PIX);
  const int n0 = (blk % (NN / P1_PIX)) * P1_PIX;
  const int t = threadIdx.x;

  for (int it = 0; it < (CC * P1_PIX) / 256; ++it) {
    int lin = t + it * 256;
    int c = lin / P1_PIX, j = lin % P1_PIX;
    size_t g = ((size_t)b * CC + c) * NN + n0 + j;
    xs[c][j] = x[g];
    cs[c][j] = cond[g];
  }
  __syncthreads();

  const int j   = t % P1_PIX;
  const int grp = t / P1_PIX;
  const int n   = n0 + j;

  for (int oo = 0; oo < 4; ++oo) {
    int o = grp * 4 + oo;
    float qa = bq[o], ka = bk[o];
    for (int c = 0; c < CC; ++c) {
      qa = fmaf(wq[o * CC + c], xs[c][j], qa);
      ka = fmaf(wk[o * CC + c], cs[c][j], ka);
    }
    Qb[((size_t)b * NN + n) * CQKD + o] = (__bf16)qa;
    Kb[((size_t)b * NN + n) * CQKD + o] = (__bf16)ka;
  }
  for (int oo = 0; oo < 32; ++oo) {
    int o = grp * 32 + oo;
    float va = bv[o];
    for (int c = 0; c < CC; ++c)
      va = fmaf(wv[o * CC + c], cs[c][j], va);
    Vb[((size_t)b * NN + n) * CC + o] = (__bf16)va;
  }
}

// ---------------- TDM: 2-D tile load via Tensor Data Mover descriptor --------
// Builds a user D# (group0: count/lds/global/type, group1: dims/strides/pad)
// and issues TENSOR_LOAD_TO_LDS. Row padding is done by the TDM itself:
//   pad_interval code c -> 2^(c+1) DWORDs, pad_amount code a -> (a+1) DWORDs.
__device__ __forceinline__ void tdm_load_2d(
    unsigned lds_addr, const void* gptr,
    unsigned tensor_d0, unsigned rows, unsigned tile_d0, unsigned stride0,
    unsigned pad_interval_code, unsigned pad_amount_code)
{
  unsigned long long ga = (unsigned long long)gptr;
  u32x4 g0;
  g0[0] = 1u;                                          // count=1 user descriptor
  g0[1] = lds_addr;                                    // LDS byte address
  g0[2] = (unsigned)(ga & 0xffffffffu);                // global_addr[31:0]
  g0[3] = (unsigned)((ga >> 32) & 0x1ffffffu) | (2u << 30);  // addr[56:32]|type=2
  u32x8 g1;
  g1[0] = (1u << 16)                                   // data_size = 2 bytes
        | (1u << 20)                                   // pad_enable
        | (pad_interval_code << 22)
        | (pad_amount_code << 25);
  g1[1] = (tensor_d0 & 0xffffu) << 16;                 // tensor_dim0[15:0]
  g1[2] = (tensor_d0 >> 16) | ((rows & 0xffffu) << 16);// dim0 hi | tensor_dim1 lo
  g1[3] = (rows >> 16) | ((tile_d0 & 0xffffu) << 16);  // dim1 hi | tile_dim0
  g1[4] = (rows & 0xffffu);                            // tile_dim1 | tile_dim2=0
  g1[5] = stride0;                                     // tensor_dim0_stride[31:0]
  g1[6] = 0u;
  g1[7] = 0u;
  asm volatile("tensor_load_to_lds %0, %1" :: "s"(g0), "s"(g1) : "memory");
}

// ---------------- Phase 2: flash attention, TDM-fed, double-buffered ---------
#define QT    64     // query rows per block
#define KT    64     // keys per inner tile
#define CPW   128    // V columns per wave col-group
#define KPAD  72     // K row pad (bf16): 144B rows, 16B-aligned, conflict-free
#define VPADC 520    // V row pad (bf16): halves land on disjoint bank sets
#define PPAD  68     // P scratch row pad

__global__ __launch_bounds__(512) void attn_kernel(
    const __bf16* __restrict__ Qb, const __bf16* __restrict__ Kb,
    const __bf16* __restrict__ Vb, const float* __restrict__ x,
    const float* __restrict__ gamma, float* __restrict__ y)
{
  __shared__ __bf16 Ks[2][KT][KPAD];      // K tiles: [key][cqk] (TDM-padded)
  __shared__ __bf16 Vs[2][KT][VPADC];     // V tiles: [key][channel] (TDM-padded)
  __shared__ __bf16 Ps[16][16][PPAD];     // per-wave P scratch / epilogue xpose

  const int b  = blockIdx.x / (NN / QT);
  const int i0 = (blockIdx.x % (NN / QT)) * QT;
  const int t    = threadIdx.x;
  const int lane = t & 31;
  const int w    = t >> 5;     // wave 0..15
  const int r    = w & 3;      // query-row group (16 rows)
  const int g    = w >> 2;     // V-column group (128 cols)
  const int m16  = lane & 15;
  const int h    = lane >> 4;

  // A-operand (16x32 bf16): lanes 0-15 = rows, dword d holds K pair at
  //   kbase(d,h) = 8h + (d<4 ? 2d : 16+2(d-4)).  B operands = B^T in A layout.
  auto kbase = [](int d, int hh) { return 8 * hh + ((d < 4) ? 2 * d : 16 + 2 * (d - 4)); };

  // Load this wave's Q rows (16 x 64) once.
  bf16x16 Qa[2];
  {
    const __bf16* qrow = Qb + ((size_t)b * NN + i0 + 16 * r + m16) * CQKD;
    #pragma unroll
    for (int ch = 0; ch < 2; ++ch) {
      union { bf16x16 v; unsigned u[8]; } tmp;
      #pragma unroll
      for (int d = 0; d < 8; ++d)
        tmp.u[d] = *(const unsigned*)(qrow + ch * 32 + kbase(d, h));
      Qa[ch] = tmp.v;
    }
  }

  f32x8 O[8];
  #pragma unroll
  for (int ct = 0; ct < 8; ++ct) O[ct] = (f32x8){0,0,0,0,0,0,0,0};
  float mrun[8], lrun[8];
  #pragma unroll
  for (int v = 0; v < 8; ++v) { mrun[v] = -3.0e38f; lrun[v] = 0.0f; }

  const __bf16* Kbase = Kb + (size_t)b * NN * CQKD;
  const __bf16* Vbase = Vb + (size_t)b * NN * CC;

  // Prologue: TDM-stage tile 0 into buffer 0 (one wave drives the DMA).
  if (w == 0) {
    tdm_load_2d((unsigned)(uintptr_t)&Ks[0][0][0], Kbase,
                CQKD, KT, CQKD, CQKD, /*32 dw*/4, /*4 dw*/3);
    tdm_load_2d((unsigned)(uintptr_t)&Vs[0][0][0], Vbase,
                CC, KT, CC, CC, /*256 dw*/7, /*4 dw*/3);
  }

  const int NT = NN / KT;
  for (int kt = 0; kt < NT; ++kt) {
    const int buf = kt & 1;
    __syncthreads();     // everyone done reading buf^1 from iteration kt-1
    if (w == 0) {
      if (kt + 1 < NT) {
        tdm_load_2d((unsigned)(uintptr_t)&Ks[buf ^ 1][0][0],
                    Kbase + (size_t)(kt + 1) * KT * CQKD,
                    CQKD, KT, CQKD, CQKD, 4, 3);
        tdm_load_2d((unsigned)(uintptr_t)&Vs[buf ^ 1][0][0],
                    Vbase + (size_t)(kt + 1) * KT * CC,
                    CC, KT, CC, CC, 7, 3);
        __builtin_amdgcn_s_wait_tensorcnt(2);  // current tile's pair complete
      } else {
        __builtin_amdgcn_s_wait_tensorcnt(0);
      }
    }
    __syncthreads();     // release: buf is fully staged

    // ---- S = Q * K^T  (16 rows x 64 keys per wave) ----
    f32x8 S[4];
    #pragma unroll
    for (int jt = 0; jt < 4; ++jt) {
      f32x8 acc = (f32x8){0,0,0,0,0,0,0,0};
      #pragma unroll
      for (int ch = 0; ch < 2; ++ch) {
        union { bf16x16 v; unsigned u[8]; } tb;
        const __bf16* krow = &Ks[buf][jt * 16 + m16][ch * 32];
        #pragma unroll
        for (int d = 0; d < 8; ++d)
          tb.u[d] = *(const unsigned*)(krow + kbase(d, h));
        acc = __builtin_amdgcn_wmma_f32_16x16x32_bf16(
            false, Qa[ch], false, tb.v, (short)0, acc, false, false);
      }
      S[jt] = acc;
    }

    // ---- online softmax (row = v + 8h lives in a 16-lane half) ----
    float scale[8];
    #pragma unroll
    for (int v = 0; v < 8; ++v) {
      float mv = mrun[v];
      #pragma unroll
      for (int jt = 0; jt < 4; ++jt) mv = fmaxf(mv, S[jt][v]);
      #pragma unroll
      for (int d = 1; d <= 8; d <<= 1) mv = fmaxf(mv, __shfl_xor(mv, d, 32));
      scale[v] = __expf(mrun[v] - mv);
      mrun[v] = mv;
      float s = 0.f;
      #pragma unroll
      for (int jt = 0; jt < 4; ++jt) {
        float p = __expf(S[jt][v] - mv);
        S[jt][v] = p;
        s += p;
      }
      #pragma unroll
      for (int d = 1; d <= 8; d <<= 1) s += __shfl_xor(s, d, 32);
      lrun[v] = lrun[v] * scale[v] + s;
    }
    #pragma unroll
    for (int ct = 0; ct < 8; ++ct)
      #pragma unroll
      for (int v = 0; v < 8; ++v) O[ct][v] *= scale[v];

    // ---- P: D-layout -> A-layout via per-wave LDS scratch ----
    #pragma unroll
    for (int jt = 0; jt < 4; ++jt)
      #pragma unroll
      for (int v = 0; v < 8; ++v)
        Ps[w][v + 8 * h][jt * 16 + m16] = (__bf16)S[jt][v];
    asm volatile("s_wait_dscnt 0" ::: "memory");

    bf16x16 Pa[2];
    #pragma unroll
    for (int ch = 0; ch < 2; ++ch) {
      union { bf16x16 v; unsigned u[8]; } tp;
      const __bf16* prow = &Ps[w][m16][ch * 32];
      #pragma unroll
      for (int d = 0; d < 8; ++d)
        tp.u[d] = *(const unsigned*)(prow + kbase(d, h));
      Pa[ch] = tp.v;
    }

    // ---- O += P * V : B^T gathered from row-major Vs with packed u16 loads --
    #pragma unroll
    for (int ct = 0; ct < 8; ++ct) {
      const int c = g * CPW + ct * 16 + m16;
      #pragma unroll
      for (int ch = 0; ch < 2; ++ch) {
        bf16x16 tb;
        #pragma unroll
        for (int d = 0; d < 8; ++d) {
          int kb = ch * 32 + kbase(d, h);
          tb[2 * d]     = Vs[buf][kb][c];
          tb[2 * d + 1] = Vs[buf][kb + 1][c];
        }
        O[ct] = __builtin_amdgcn_wmma_f32_16x16x32_bf16(
            false, Pa[ch], false, tb, (short)0, O[ct], false, false);
      }
    }
  }

  // ---- epilogue: normalize, transpose 16x16 tiles through LDS, fuse residual
  const float gam = gamma[0];
  float rinv[8];
  #pragma unroll
  for (int v = 0; v < 8; ++v) rinv[v] = 1.0f / lrun[v];
  float* tw = (float*)&Ps[w][0][0];       // 16*17*4 B fits in 2176 B region
  const int i = i0 + 16 * r + m16;
  for (int ct = 0; ct < 8; ++ct) {
    #pragma unroll
    for (int v = 0; v < 8; ++v)
      tw[(v + 8 * h) * 17 + m16] = O[ct][v] * rinv[v];
    asm volatile("s_wait_dscnt 0" ::: "memory");
    #pragma unroll
    for (int cp = 0; cp < 8; ++cp) {
      int col = 2 * cp + h;
      int c = g * CPW + ct * 16 + col;
      float val = tw[m16 * 17 + col];     // transposed read: lane -> query row
      size_t idx = ((size_t)b * CC + c) * NN + i;
      y[idx] = gam * val + x[idx];
    }
    asm volatile("s_wait_dscnt 0" ::: "memory");
  }
}

extern "C" void kernel_launch(void* const* d_in, const int* in_sizes, int n_in,
                              void* d_out, int out_size, void* d_ws, size_t ws_size,
                              hipStream_t stream) {
  const float* x     = (const float*)d_in[0];
  const float* cond  = (const float*)d_in[1];
  const float* wq    = (const float*)d_in[2];
  const float* bq    = (const float*)d_in[3];
  const float* wk    = (const float*)d_in[4];
  const float* bk    = (const float*)d_in[5];
  const float* wv    = (const float*)d_in[6];
  const float* bv    = (const float*)d_in[7];
  const float* gamma = (const float*)d_in[8];
  float* y = (float*)d_out;

  char* ws = (char*)d_ws;
  __bf16* Qb = (__bf16*)(ws);
  __bf16* Kb = (__bf16*)(ws + (size_t)BB * NN * CQKD * 2);
  __bf16* Vb = (__bf16*)(ws + (size_t)BB * NN * CQKD * 4);

  proj_kernel<<<dim3(BB * (NN / P1_PIX)), dim3(256), 0, stream>>>(
      x, cond, wq, bq, wk, bk, wv, bv, Qb, Kb, Vb);
  attn_kernel<<<dim3(BB * (NN / QT)), dim3(512), 0, stream>>>(
      Qb, Kb, Vb, x, gamma, y);
}
